// MapEncoderPtsMA_8186207666359
// MI455X (gfx1250) — matrix operand order repaired and checked by reading, not compiled, not run on hardware
//
#include <hip/hip_runtime.h>
#include <hip/hip_bf16.h>

// ---------------------------------------------------------------------------
// MapEncoder (single-seed MHA pooling + FFN) for MI455X / gfx1250, wave32.
//
// Roofline: ~15MB of HBM traffic (L2-resident working set) vs ~5.9 GFLOP after
// algebraic folding (q is one shared seed -> fold wk@q into a 128x8 score
// matrix; pool attention before wv). Compute-bound => GEMM-shaped work
// (scores, pooling, FFN) goes through v_wmma_f32_16x16x32_f16 (f32 accum).
// The FFN input tile is staged into LDS by the Tensor Data Mover
// (tensor_load_to_lds + s_wait_tensorcnt) when the builtin is available.
// ---------------------------------------------------------------------------

typedef _Float16 h16;
typedef __attribute__((ext_vector_type(16))) _Float16 v16h;
typedef __attribute__((ext_vector_type(8)))  float    v8f;
typedef __attribute__((ext_vector_type(4)))  unsigned int u32x4;
typedef __attribute__((ext_vector_type(8)))  int          i32x8;
typedef __attribute__((ext_vector_type(4)))  int          i32x4;

#if __has_builtin(__builtin_amdgcn_tensor_load_to_lds) && \
    __has_builtin(__builtin_amdgcn_s_wait_tensorcnt)
#define USE_TDM 1
#else
#define USE_TDM 0
#endif

constexpr int Bc = 16, Mc = 8, Sc = 100, Pc = 40, Dc = 128, Dff = 384;
constexpr int Nc = Bc * Mc * Sc;             // 12800 segments
constexpr int OUT_ELEMS = Sc * Bc * Mc * Dc; // 1,638,400 floats

__device__ __forceinline__ v8f wmma_f16(v16h a, v16h b, v8f c) {
  // 8 args: (neg_a, A, neg_b, B, c_mod, C, reuse_a, reuse_b)
  return __builtin_amdgcn_wmma_f32_16x16x32_f16(false, a, false, b, (short)0, c,
                                                false, false);
}

__device__ __forceinline__ float wred_sum32(float x) {
#pragma unroll
  for (int m = 1; m < 32; m <<= 1) x += __shfl_xor(x, m, 32);
  return x;
}

// ---------------------------------------------------------------------------
// Kernel 0: one-time precompute.
//   q = seeds @ wq + bq
//   kq_t[n][k] = 0.25 * sum_d wk[k, 16n+d] * q[16n+d]   (n-major f16, n<8 real)
//   w1t[n][k] = w1[k][n] (f16)   w2t[n][k] = w2[k][n] (f16)
//   wv_t[o][d] = wv[d][o] (f16)  ow_t[o][d] = out_w[d][o] (f16)
// n-major layouts make WMMA B-fragment (k, k+1) pairs contiguous; transposed
// f16 wv/out_w halve kernel-2's dominant L2 stream and feed v_fma_mix_f32.
// ---------------------------------------------------------------------------
__global__ __launch_bounds__(256) void precompute_kernel(
    const float* __restrict__ seeds, const float* __restrict__ wq,
    const float* __restrict__ bq, const float* __restrict__ wk,
    const float* __restrict__ w1, const float* __restrict__ w2,
    const float* __restrict__ wv, const float* __restrict__ out_w,
    h16* __restrict__ kq_t, h16* __restrict__ w1t, h16* __restrict__ w2t,
    h16* __restrict__ wv_t, h16* __restrict__ ow_t) {
  __shared__ float sQ[Dc];
  const int tid = threadIdx.x;
  if (tid < Dc) {
    float acc = bq[tid];
    for (int d = 0; d < Dc; ++d) acc += seeds[d] * wq[d * Dc + tid];
    sQ[tid] = acc;
  }
  __syncthreads();
  for (int i = tid; i < 16 * Dc; i += 256) {
    const int n = i >> 7, k = i & 127;
    float v = 0.0f;
    if (n < 8) {
      for (int d = 0; d < 16; ++d) v += wk[k * Dc + n * 16 + d] * sQ[n * 16 + d];
      v *= 0.25f;  // HD^-0.5 folded into the score matrix
    }
    kq_t[i] = (h16)v;
  }
  for (int i = tid; i < Dff * Dc; i += 256) {  // w1t[n][k]
    const int n = i >> 7, k = i & 127;
    w1t[i] = (h16)w1[k * Dff + n];
  }
  for (int i = tid; i < Dc * Dff; i += 256) {  // w2t[n][k]
    const int n = i / Dff, k = i % Dff;
    w2t[i] = (h16)w2[k * Dc + n];
  }
  for (int i = tid; i < Dc * Dc; i += 256) {  // transposed f16 wv / out_w
    const int o = i >> 7, d = i & 127;
    wv_t[i] = (h16)wv[d * Dc + o];
    ow_t[i] = (h16)out_w[d * Dc + o];
  }
}

// ---------------------------------------------------------------------------
// Kernel 1: seg_mask (B,M,S) with the "all segments masked -> unmask s=0" fix.
// ---------------------------------------------------------------------------
__global__ __launch_bounds__(128) void seg_mask_kernel(
    const float* __restrict__ roads, float* __restrict__ mask_out) {
  __shared__ int sAny;
  const int bm = blockIdx.x;
  const int tid = threadIdx.x;
  if (tid == 0) sAny = 0;
  __syncthreads();
  int masked = 1;
  if (tid < Sc) {
    const float* rd = roads + ((size_t)bm * Sc + tid) * Pc * 4;
    float cnt = 0.0f;
    for (int p = 0; p < Pc; ++p) cnt += rd[p * 4 + 3];
    masked = (cnt == 0.0f) ? 1 : 0;
    if (!masked) atomicOr(&sAny, 1);
  }
  __syncthreads();
  if (tid < Sc) {
    int m = masked;
    if (sAny == 0 && tid == 0) m = 0;  // unmask seg 0 of fully-empty (b,m)
    mask_out[(size_t)bm * Sc + tid] = (float)m;
  }
}

// ---------------------------------------------------------------------------
// Kernel 2: per-segment attention pooling (1 wave / segment).
//   feats = attrs @ lin_w + lin_b            (VALU, K=3)
//   scores = feats(48x128) @ kq_t(128x16)    -> 12 WMMAs
//   masked softmax over P (cross-half via shfl_xor 16)
//   pooled = attn(16x64) @ featsT(64x128)    -> 16 WMMAs
//   ctx = pooled_h @ wv[:,h-block] + bv; emb = LN1(ctx @ out_w + out_b)
// ---------------------------------------------------------------------------
__global__ __launch_bounds__(32) void map_seg_attn_kernel(
    const float* __restrict__ roads, const float* __restrict__ lin_w,
    const float* __restrict__ lin_b, const h16* __restrict__ kq_t,
    const h16* __restrict__ wv_t, const float* __restrict__ bv,
    const h16* __restrict__ ow_t, const float* __restrict__ out_b,
    const float* __restrict__ ln1_g, const float* __restrict__ ln1_b,
    float* __restrict__ emb_out) {
  __shared__ float sAttr[48][4];
  __shared__ float sLinW[4][Dc];
  __shared__ float sValid[48];
  __shared__ h16 sFeatR[48 * Dc];   // row-major, WMMA A source
  __shared__ h16 sFeatC[Dc * 64];   // col-major (p padded to 64), WMMA B source
  __shared__ h16 sAttn[16 * 64];
  __shared__ float sPooled[8 * Dc];
  __shared__ float sCtx[Dc];

  const int seg = blockIdx.x;
  const int lane = threadIdx.x;
  const int hlf = lane >> 4;
  const int l15 = lane & 15;
  const float* rd = roads + (size_t)seg * (Pc * 4);

  float lsum = 0.0f;
  for (int p = lane; p < 48; p += 32) {
    float a0 = 0, a1 = 0, a2 = 0, lv = 0;
    if (p < Pc) { a0 = rd[p*4]; a1 = rd[p*4+1]; a2 = rd[p*4+2]; lv = rd[p*4+3]; }
    sAttr[p][0] = a0; sAttr[p][1] = a1; sAttr[p][2] = a2; sAttr[p][3] = lv;
    sValid[p] = (lv != 0.0f) ? 1.0f : 0.0f;
    lsum += lv;
  }
  for (int i = lane; i < 4 * Dc; i += 32) {
    const int r = i >> 7, c = i & 127;
    sLinW[r][c] = (r < 3) ? lin_w[r * Dc + c] : lin_b[c];
  }
  const float tot = wred_sum32(lsum);
  if (tot == 0.0f && lane == 0) sValid[0] = 1.0f;  // unmask pt0 of empty seg
  __syncthreads();

  for (int i = lane; i < 48 * Dc; i += 32) {
    const int p = i >> 7, d = i & 127;
    float f = 0.0f;
    if (p < Pc)
      f = sAttr[p][0] * sLinW[0][d] + sAttr[p][1] * sLinW[1][d] +
          sAttr[p][2] * sLinW[2][d] + sLinW[3][d];
    const h16 hf = (h16)f;
    sFeatR[p * Dc + d] = hf;
    sFeatC[d * 64 + p] = hf;
  }
  for (int i = lane; i < Dc * 16; i += 32) {  // zero pad p = 48..63
    const int d = i >> 4, p = 48 + (i & 15);
    sFeatC[d * 64 + p] = (h16)0.0f;
  }
  for (int i = lane; i < 16 * 64; i += 32) sAttn[i] = (h16)0.0f;
  __syncthreads();

  // ---- scores: 3 M-tiles x 4 K-steps -----------------------------------
  float sc[3][8];
#pragma unroll
  for (int t = 0; t < 3; ++t) {
    v8f c = {};
    const int row = t * 16 + l15;
#pragma unroll
    for (int ks = 0; ks < 4; ++ks) {
      v16h a, b;
#pragma unroll
      for (int v = 0; v < 8; ++v) {
        const int ka = ks * 32 + (v >> 2) * 16 + hlf * 8 + (v & 3) * 2;
        a[2 * v] = sFeatR[row * Dc + ka];
        a[2 * v + 1] = sFeatR[row * Dc + ka + 1];
        const int kb = ks * 32 + (v >> 2) * 16 + hlf * 8 + (v & 3) * 2;
        b[2 * v] = kq_t[l15 * Dc + kb];
        b[2 * v + 1] = kq_t[l15 * Dc + kb + 1];
      }
      c = wmma_f16(a, b, c);
    }
#pragma unroll
    for (int v = 0; v < 8; ++v) sc[t][v] = c[v];
  }

  // ---- masked softmax (head = l15; p split across xor-16 lane pairs) ----
  float mx = -3.0e38f;
#pragma unroll
  for (int t = 0; t < 3; ++t)
#pragma unroll
    for (int v = 0; v < 8; ++v) {
      const int p = t * 16 + hlf * 8 + v;
      const float s = (sValid[p] != 0.0f) ? sc[t][v] : -1.0e9f;
      sc[t][v] = s;
      mx = fmaxf(mx, s);
    }
  mx = fmaxf(mx, __shfl_xor(mx, 16, 32));
  float se = 0.0f;
#pragma unroll
  for (int t = 0; t < 3; ++t)
#pragma unroll
    for (int v = 0; v < 8; ++v) {
      const float e = __expf(sc[t][v] - mx);
      sc[t][v] = e;
      se += e;
    }
  se += __shfl_xor(se, 16, 32);
  const float inv = 1.0f / se;
  if (l15 < 8) {
#pragma unroll
    for (int t = 0; t < 3; ++t)
#pragma unroll
      for (int v = 0; v < 8; ++v) {
        const int p = t * 16 + hlf * 8 + v;
        sAttn[l15 * 64 + p] = (h16)(sc[t][v] * inv);
      }
  }
  __syncthreads();

  // ---- pooled = attn @ feats : 8 N-tiles x 2 K-steps --------------------
  for (int nt = 0; nt < 8; ++nt) {
    v8f c = {};
    const int dcol = nt * 16 + l15;
#pragma unroll
    for (int ks = 0; ks < 2; ++ks) {
      v16h a, b;
#pragma unroll
      for (int v = 0; v < 8; ++v) {
        const int ka = ks * 32 + (v >> 2) * 16 + hlf * 8 + (v & 3) * 2;
        a[2 * v] = sAttn[l15 * 64 + ka];
        a[2 * v + 1] = sAttn[l15 * 64 + ka + 1];
        const int kb = ks * 32 + (v >> 2) * 16 + hlf * 8 + (v & 3) * 2;
        b[2 * v] = sFeatC[dcol * 64 + kb];
        b[2 * v + 1] = sFeatC[dcol * 64 + kb + 1];
      }
      c = wmma_f16(a, b, c);
    }
    if (hlf == 0) {  // rows M=0..7 are the real heads
#pragma unroll
      for (int v = 0; v < 8; ++v) sPooled[v * Dc + dcol] = c[v];
    }
  }
  __syncthreads();

  // ---- ctx = pooled_h @ wv[:, h-block] + bv (f16 weights, fma_mix) ------
  for (int o = lane; o < Dc; o += 32) {
    const int h = o >> 4;
    float acc = bv[o];
    const h16* wr = wv_t + (size_t)o * Dc;
    for (int d = 0; d < Dc; ++d) acc += sPooled[h * Dc + d] * (float)wr[d];
    sCtx[o] = acc;
  }
  __syncthreads();

  // ---- emb = LN1(ctx @ out_w + out_b) -----------------------------------
  float vals[4];
  float s1 = 0.0f, s2 = 0.0f;
#pragma unroll
  for (int i = 0; i < 4; ++i) {
    const int o = lane + i * 32;
    float acc = out_b[o];
    const h16* wr = ow_t + (size_t)o * Dc;
    for (int d = 0; d < Dc; ++d) acc += sCtx[d] * (float)wr[d];
    vals[i] = acc;
    s1 += acc;
    s2 += acc * acc;
  }
  s1 = wred_sum32(s1);
  s2 = wred_sum32(s2);
  const float mean = s1 * (1.0f / 128.0f);
  const float var = s2 * (1.0f / 128.0f) - mean * mean;
  const float rs = rsqrtf(var + 1e-5f);
#pragma unroll
  for (int i = 0; i < 4; ++i) {
    const int o = lane + i * 32;
    emb_out[(size_t)seg * Dc + o] = (vals[i] - mean) * rs * ln1_g[o] + ln1_b[o];
  }
}

// ---------------------------------------------------------------------------
// Kernel 3: batched FFN + residual + LN2 + transpose, 16 segments per block.
//   emb tile (16x128 f32, 8KB) staged into LDS by the Tensor Data Mover.
//   h1 = relu(emb@w1+b1) : 24 N-tiles x 4 K-steps (96 WMMAs)
//   y  = emb + h1@w2+b2  :  8 N-tiles x 12 K-steps (96 WMMAs)
// ---------------------------------------------------------------------------
__global__ __launch_bounds__(128) void ffn_ln_out_kernel(
    const float* __restrict__ emb, const h16* __restrict__ w1t,
    const float* __restrict__ b1, const h16* __restrict__ w2t,
    const float* __restrict__ b2, const float* __restrict__ ln2_g,
    const float* __restrict__ ln2_b, float* __restrict__ out) {
  __shared__ h16 sEmbH[16 * Dc];
  __shared__ float sEmbF[16 * Dc];
  __shared__ h16 sH1[16 * 392];  // 384 + 8 pad halfs per row
  __shared__ float sY[16 * Dc];

  const int tid = threadIdx.x;
  const int wave = tid >> 5;
  const int lane = tid & 31;
  const int hlf = lane >> 4;
  const int l15 = lane & 15;
  const int base = blockIdx.x * 16;

  // Warm the per-wave weight slabs (global_prefetch_b8).
  __builtin_prefetch(w1t + (size_t)(wave * 6 * 16) * Dc, 0, 3);
  __builtin_prefetch(w2t + (size_t)(wave * 2 * 16) * Dff, 0, 3);

#if USE_TDM
  if (tid < 32) {  // wave 0 issues one TDM descriptor for the 16x128 f32 tile
    const unsigned lds_off = (unsigned)(uintptr_t)(&sEmbF[0]);
    const unsigned long long ga =
        (unsigned long long)(uintptr_t)(emb + (size_t)base * Dc);
    u32x4 g0;
    g0[0] = 1u;                                   // count=1, user descriptor
    g0[1] = lds_off;                              // lds_addr (bytes)
    g0[2] = (unsigned)(ga & 0xffffffffu);         // global_addr[31:0]
    g0[3] = (unsigned)((ga >> 32) & 0x01ffffffu)  // global_addr[56:32]
            | (2u << 30);                         // type=2 ("image")
    i32x8 g1;
    g1[0] = (2 << 16);    // workgroup_mask=0 | data_size=2 (4B elements)
    g1[1] = (128 << 16);  // tensor_dim0 = 128 (low 16 bits at [31:16])
    g1[2] = (16 << 16);   // tensor_dim0 hi=0 | tensor_dim1 = 16
    g1[3] = (128 << 16);  // tensor_dim1 hi=0 | tile_dim0 = 128
    g1[4] = 16;           // tile_dim1 = 16 | tile_dim2 = 0
    g1[5] = 128;          // tensor_dim0_stride = 128 elements
    g1[6] = 0;
    g1[7] = 0;
    i32x4 g2 = {0, 0, 0, 0}, g3 = {0, 0, 0, 0};
#if defined(__clang_major__) && __clang_major__ >= 23
    i32x8 g4 = {0, 0, 0, 0, 0, 0, 0, 0};
    __builtin_amdgcn_tensor_load_to_lds(g0, g1, g2, g3, g4, 0);
#else
    __builtin_amdgcn_tensor_load_to_lds(g0, g1, g2, g3, 0);
#endif
    __builtin_amdgcn_s_wait_tensorcnt(0);
  }
  __syncthreads();
  for (int i = tid; i < 16 * Dc; i += 128) sEmbH[i] = (h16)sEmbF[i];
  __syncthreads();
#else
  for (int i = tid; i < 16 * Dc; i += 128) {
    const float v = emb[(size_t)base * Dc + i];
    sEmbF[i] = v;
    sEmbH[i] = (h16)v;
  }
  __syncthreads();
#endif

  v16h afr[4];  // A fragments shared across all N-tiles of GEMM1
#pragma unroll
  for (int ks = 0; ks < 4; ++ks) {
#pragma unroll
    for (int v = 0; v < 8; ++v) {
      const int ka = ks * 32 + (v >> 2) * 16 + hlf * 8 + (v & 3) * 2;
      afr[ks][2 * v] = sEmbH[l15 * Dc + ka];
      afr[ks][2 * v + 1] = sEmbH[l15 * Dc + ka + 1];
    }
  }

  for (int j = 0; j < 6; ++j) {  // GEMM1: wave handles 6 of 24 N-tiles
    const int col = (wave * 6 + j) * 16 + l15;
    const float bias = b1[col];
    v8f c = {bias, bias, bias, bias, bias, bias, bias, bias};
#pragma unroll
    for (int ks = 0; ks < 4; ++ks) {
      v16h b;
#pragma unroll
      for (int v = 0; v < 8; ++v) {
        const int kb = ks * 32 + (v >> 2) * 16 + hlf * 8 + (v & 3) * 2;
        b[2 * v] = w1t[(size_t)col * Dc + kb];
        b[2 * v + 1] = w1t[(size_t)col * Dc + kb + 1];
      }
      c = wmma_f16(afr[ks], b, c);
    }
#pragma unroll
    for (int v = 0; v < 8; ++v) {
      const int row = v + hlf * 8;
      sH1[row * 392 + col] = (h16)fmaxf(c[v], 0.0f);  // ReLU
    }
  }
  __syncthreads();

  for (int j = 0; j < 2; ++j) {  // GEMM2: wave handles 2 of 8 N-tiles
    const int col = (wave * 2 + j) * 16 + l15;
    const float bias = b2[col];
    v8f c = {bias, bias, bias, bias, bias, bias, bias, bias};
    for (int ks = 0; ks < 12; ++ks) {
      v16h a, b;
#pragma unroll
      for (int v = 0; v < 8; ++v) {
        const int ka = ks * 32 + (v >> 2) * 16 + hlf * 8 + (v & 3) * 2;
        a[2 * v] = sH1[l15 * 392 + ka];
        a[2 * v + 1] = sH1[l15 * 392 + ka + 1];
        const int kb = ks * 32 + (v >> 2) * 16 + hlf * 8 + (v & 3) * 2;
        b[2 * v] = w2t[(size_t)col * Dff + kb];
        b[2 * v + 1] = w2t[(size_t)col * Dff + kb + 1];
      }
      c = wmma_f16(a, b, c);
    }
#pragma unroll
    for (int v = 0; v < 8; ++v) {
      const int row = v + hlf * 8;
      sY[row * Dc + col] = c[v] + sEmbF[row * Dc + col];  // residual
    }
  }
  __syncthreads();

  // LN2 + scatter to (S, B, M, D)
  {
    const int r = tid >> 3;   // segment row within tile
    const int g = tid & 7;    // 16-wide column group
    float vals[16];
    float s1 = 0.0f, s2 = 0.0f;
#pragma unroll
    for (int i = 0; i < 16; ++i) {
      const float v = sY[r * Dc + g * 16 + i];
      vals[i] = v;
      s1 += v;
      s2 += v * v;
    }
#pragma unroll
    for (int m = 1; m < 8; m <<= 1) {
      s1 += __shfl_xor(s1, m, 32);
      s2 += __shfl_xor(s2, m, 32);
    }
    const float mean = s1 * (1.0f / 128.0f);
    const float var = s2 * (1.0f / 128.0f) - mean * mean;
    const float rs = rsqrtf(var + 1e-5f);
    const int n = base + r;
    const int s = n % Sc;
    const int bm = n / Sc;
    const size_t ob = ((size_t)s * (Bc * Mc) + bm) * Dc;
#pragma unroll
    for (int i = 0; i < 16; ++i) {
      const int d = g * 16 + i;
      out[ob + d] = (vals[i] - mean) * rs * ln2_g[d] + ln2_b[d];
    }
  }
}

// ---------------------------------------------------------------------------
extern "C" void kernel_launch(void* const* d_in, const int* in_sizes, int n_in,
                              void* d_out, int out_size, void* d_ws,
                              size_t ws_size, hipStream_t stream) {
  (void)in_sizes; (void)n_in; (void)out_size; (void)ws_size;
  const float* roads = (const float*)d_in[0];
  // d_in[1] agents_emb: unused by the reference computation
  const float* seeds = (const float*)d_in[2];
  const float* lin_w = (const float*)d_in[3];
  const float* lin_b = (const float*)d_in[4];
  const float* wq = (const float*)d_in[5];
  const float* wk = (const float*)d_in[6];
  const float* wv = (const float*)d_in[7];
  const float* bq = (const float*)d_in[8];
  // d_in[9] bk: adds a per-head constant to scores -> softmax-invariant, dropped
  const float* bv = (const float*)d_in[10];
  const float* out_w = (const float*)d_in[11];
  const float* out_b = (const float*)d_in[12];
  const float* ln1_g = (const float*)d_in[13];
  const float* ln1_b = (const float*)d_in[14];
  const float* w1 = (const float*)d_in[15];
  const float* b1 = (const float*)d_in[16];
  const float* w2 = (const float*)d_in[17];
  const float* b2 = (const float*)d_in[18];
  const float* ln2_g = (const float*)d_in[19];
  const float* ln2_b = (const float*)d_in[20];

  char* ws = (char*)d_ws;
  h16* kq_t = (h16*)(ws + 0);           //  16*128 f16 =   4 KB
  h16* w1t = (h16*)(ws + 4096);         // 384*128 f16 =  96 KB
  h16* w2t = (h16*)(ws + 102400);       // 128*384 f16 =  96 KB
  h16* wv_t = (h16*)(ws + 200704);      // 128*128 f16 =  32 KB
  h16* ow_t = (h16*)(ws + 233472);      // 128*128 f16 =  32 KB
  float* embw = (float*)(ws + 266240);  // 12800*128 f32 = 6.25 MB

  float* out = (float*)d_out;
  float* mask_out = out + OUT_ELEMS;  // seg_mask (B,M,S) as 0/1 floats

  precompute_kernel<<<1, 256, 0, stream>>>(seeds, wq, bq, wk, w1, w2, wv, out_w,
                                           kq_t, w1t, w2t, wv_t, ow_t);
  seg_mask_kernel<<<Bc * Mc, 128, 0, stream>>>(roads, mask_out);
  map_seg_attn_kernel<<<Nc, 32, 0, stream>>>(roads, lin_w, lin_b, kq_t, wv_t,
                                             bv, ow_t, out_b, ln1_g, ln1_b,
                                             embw);
  ffn_ln_out_kernel<<<Nc / 16, 128, 0, stream>>>(embw, w1t, b1, w2t, b2, ln2_g,
                                                 ln2_b, out);
}
